// ModulatedAttLayer_90683939488097
// MI455X (gfx1250) — matrix operand order repaired and checked by reading, not compiled, not run on hardware
//
#include <hip/hip_runtime.h>

// ModulatedAttLayer on MI455X (gfx1250, wave32, WMMA).
// ~215 GFLOP total; all large contractions run on v_wmma_f32_16x16x32_bf16
// with f32 accumulate. Round-2 change: 64x64 per-wave register blocking on the
// two big GEMMs (k_proj, k_mask_out) -> 16 WMMA per 8KB of fragment loads
// (~32 FLOP/B from L2) instead of 4 WMMA per 10KB (~16 FLOP/B), cutting L2
// fragment traffic ~4x so the matrix pipes, not L2 bandwidth, set the ceiling.

#define B_     256
#define C_     2048
#define INTER_ 1024
#define N_     49
#define NP_    64
#define KFLAT_ (C_ * N_)   // 100352

typedef __attribute__((ext_vector_type(16))) __bf16 v16bf;
typedef __attribute__((ext_vector_type(8)))  float  v8f;

struct alignas(16) U4 { unsigned int x, y, z, w; };
struct alignas(16) F4 { float x, y, z, w; };

union Frag {
    U4 q[2];
    unsigned short s[16];
    v16bf v;
};

__device__ __forceinline__ unsigned short f32_to_bf16(float x) {
    unsigned int u = __float_as_uint(x);
    u += 0x7FFFu + ((u >> 16) & 1u);   // round-to-nearest-even
    return (unsigned short)(u >> 16);
}

__device__ __forceinline__ v8f wmma_bf16(v16bf a, v16bf b, v8f c) {
    return __builtin_amdgcn_wmma_f32_16x16x32_bf16(
        /*neg_a=*/false, a, /*neg_b=*/false, b,
        /*c_mod=*/(short)0, c, /*reuse_a=*/false, /*reuse_b=*/false);
}

// A fragment, 16x32 bf16, rows contiguous over K (contiguous K order, same as the B loaders):
// lane l: row = l&15, khalf = l>>4; elem j -> K = j + 16*khalf
__device__ __forceinline__ v16bf load_a_bf16(const unsigned short* base, int ld, int k0) {
    const int lane = threadIdx.x & 31;
    const int row = lane & 15, kh = lane >> 4;
    const unsigned short* p = base + (size_t)row * ld + k0 + kh * 16;
    Frag f;
    f.q[0] = *(const U4*)(p);        // K = 16*kh + 0..7
    f.q[1] = *(const U4*)(p + 8);    // K = 16*kh + 8..15
    return f.v;
}

// B fragment, 32x16 bf16, "NT": per-column data contiguous over K:
// lane l: col = l&15, khalf = l>>4; elem j -> K = j + 16*khalf
__device__ __forceinline__ v16bf load_b_nt_bf16(const unsigned short* base, int ld, int k0) {
    const int lane = threadIdx.x & 31;
    const int col = lane & 15, kh = lane >> 4;
    const unsigned short* p = base + (size_t)col * ld + k0 + kh * 16;
    Frag f;
    f.q[0] = *(const U4*)(p);
    f.q[1] = *(const U4*)(p + 8);
    return f.v;
}

// B fragment where K advances with stride ld (element (k, col) = base[col + k*ld])
__device__ __forceinline__ v16bf load_b_strided_bf16(const unsigned short* base, int ld, int k0) {
    const int lane = threadIdx.x & 31;
    const int col = lane & 15, kh = lane >> 4;
    const unsigned short* p = base + col + (size_t)(k0 + kh * 16) * ld;
    Frag f;
#pragma unroll
    for (int j = 0; j < 16; ++j) f.s[j] = p[(size_t)j * ld];
    return f.v;
}

// A fragment gathered from f32 memory with inline bf16 conversion
__device__ __forceinline__ v16bf load_a_f32(const float* base, int ld, int k0) {
    const int lane = threadIdx.x & 31;
    const int row = lane & 15, kh = lane >> 4;
    const float* p = base + (size_t)row * ld + k0 + kh * 16;
    float t[16];
    *(F4*)&t[0]  = *(const F4*)(p);
    *(F4*)&t[4]  = *(const F4*)(p + 4);
    *(F4*)&t[8]  = *(const F4*)(p + 8);
    *(F4*)&t[12] = *(const F4*)(p + 12);
    Frag f;
#pragma unroll
    for (int j = 0; j < 16; ++j) f.s[j] = f32_to_bf16(t[j]);
    return f.v;
}

// B NT fragment from f32 memory with column bound (zero-pad cols >= colMax)
__device__ __forceinline__ v16bf load_b_nt_f32_guard(const float* base, size_t ld, int k0,
                                                     int colBase, int colMax) {
    const int lane = threadIdx.x & 31;
    const int col = (lane & 15) + colBase, kh = lane >> 4;
    Frag f;
    if (col < colMax) {
        const float* p = base + (size_t)col * ld + k0 + kh * 16;
        float t[16];
        *(F4*)&t[0]  = *(const F4*)(p);
        *(F4*)&t[4]  = *(const F4*)(p + 4);
        *(F4*)&t[8]  = *(const F4*)(p + 8);
        *(F4*)&t[12] = *(const F4*)(p + 12);
#pragma unroll
        for (int j = 0; j < 16; ++j) f.s[j] = f32_to_bf16(t[j]);
    } else {
#pragma unroll
        for (int j = 0; j < 16; ++j) f.s[j] = 0;
    }
    return f.v;
}

// Epilogue for a 16x64 tile: transpose through LDS, b128 row-major bf16 stores.
// Intra-wave only -> s_wait_dscnt 0, no barrier needed.
__device__ __forceinline__ void epilogue_store_bf16(unsigned short (*st)[64],
                                                    const v8f* acc, const float* b4,
                                                    unsigned short* dst, int ld) {
    const int lane = threadIdx.x & 31;
    const int col = lane & 15, hi = lane >> 4;
#pragma unroll
    for (int s = 0; s < 4; ++s) {
        const float bv = b4 ? b4[s] : 0.f;
#pragma unroll
        for (int v = 0; v < 8; ++v)
            st[v + 8 * hi][s * 16 + col] = f32_to_bf16(acc[s][v] + bv);
    }
    asm volatile("s_wait_dscnt 0" ::: "memory");
    const int r = lane >> 1, h = lane & 1;
    const U4* src = (const U4*)&st[r][h * 32];
    U4* d = (U4*)(dst + (size_t)r * ld + h * 32);
    d[0] = src[0]; d[1] = src[1]; d[2] = src[2]; d[3] = src[3];
}

// ---------------- kernels ----------------

__global__ void k_convert(const float* __restrict__ src, unsigned short* __restrict__ dst, int n) {
    int i = blockIdx.x * blockDim.x + threadIdx.x;
    if (i < n) dst[i] = f32_to_bf16(src[i]);
}

// x[b][c][n] f32 -> xT[b][n(pad64)][c] bf16 (LDS tile transpose, coalesced both sides)
__global__ void k_transpose_x(const float* __restrict__ x, unsigned short* __restrict__ xT) {
    __shared__ float tile[64][65];
    const int c0 = blockIdx.x * 64;
    const int b  = blockIdx.y;
    for (int idx = threadIdx.x; idx < 64 * 64; idx += 256) {
        const int n = idx & 63, cc = idx >> 6;
        float v = 0.f;
        if (n < N_) v = x[((size_t)b * C_ + c0 + cc) * N_ + n];
        tile[cc][n] = v;
    }
    __syncthreads();
    for (int idx = threadIdx.x; idx < 64 * 64; idx += 256) {
        const int cc = idx & 63, n = idx >> 6;
        xT[((size_t)b * NP_ + n) * C_ + c0 + cc] = f32_to_bf16(tile[cc][n]);
    }
}

// Projections: proj[w][b][n][o] = xT[b][n][:] . w[o][:] + bias[o]
// 64x64 per-wave register block: 4 A-frags (all padded-N rows) x 4 B-frags,
// 16 accumulators, 16 WMMA per 32-K step.
__global__ void __launch_bounds__(128) k_proj(const unsigned short* __restrict__ xT,
                                              const unsigned short* __restrict__ w3,
                                              const float* __restrict__ g_b,
                                              const float* __restrict__ th_b,
                                              const float* __restrict__ ph_b,
                                              unsigned short* __restrict__ proj) {
    __shared__ __align__(16) unsigned short stile[4][64][64];   // 32 KB
    const int w    = threadIdx.x >> 5;
    const int wg   = blockIdx.x * 4 + w;
    const int cg   = wg & 15;                  // 16 col groups of 64
    const int wsel = (wg >> 4) % 3;
    const int b    = wg / 48;
    const unsigned short* A  = xT + (size_t)b * NP_ * C_;
    const unsigned short* Bm = w3 + ((size_t)wsel * INTER_ + cg * 64) * C_;
    v8f acc[4][4] = {};
    for (int k = 0; k < C_; k += 32) {
        v16bf a[4];
#pragma unroll
        for (int r = 0; r < 4; ++r) a[r] = load_a_bf16(A + (size_t)r * 16 * C_, C_, k);
#pragma unroll
        for (int s = 0; s < 4; ++s) {
            const v16bf bf = load_b_nt_bf16(Bm + (size_t)s * 16 * C_, C_, k);
#pragma unroll
            for (int r = 0; r < 4; ++r)
                acc[r][s] = wmma_bf16(a[r], bf, acc[r][s]);
        }
    }
    const float* bias = (wsel == 0) ? g_b : ((wsel == 1) ? th_b : ph_b);
    const int lane = threadIdx.x & 31;
    const int col = lane & 15, hi = lane >> 4;
    unsigned short (*st)[64] = stile[w];
#pragma unroll
    for (int s = 0; s < 4; ++s) {
        const float bv = bias[cg * 64 + s * 16 + col];
#pragma unroll
        for (int r = 0; r < 4; ++r)
#pragma unroll
            for (int v = 0; v < 8; ++v)
                st[r * 16 + v + 8 * hi][s * 16 + col] = f32_to_bf16(acc[r][s][v] + bv);
    }
    asm volatile("s_wait_dscnt 0" ::: "memory");
    unsigned short* dst = proj + (((size_t)wsel * B_ + b) * NP_) * INTER_ + cg * 64;
    for (int r = lane; r < 64; r += 32) {
        const U4* src = (const U4*)&st[r][0];
        U4* d = (U4*)(dst + (size_t)r * INTER_);
#pragma unroll
        for (int q = 0; q < 8; ++q) d[q] = src[q];
    }
}

// scores[n][m] = theta[n][:] . phi[m][:]  (NT), then row softmax over m<49 -> attn bf16
__global__ void __launch_bounds__(128) k_scores_softmax(const unsigned short* __restrict__ proj,
                                                        unsigned short* __restrict__ attn) {
    __shared__ float sc[64][65];
    const int b = blockIdx.x;
    const int w = threadIdx.x >> 5;
    const unsigned short* theta = proj + ((size_t)(1 * B_ + b)) * NP_ * INTER_;
    const unsigned short* phi   = proj + ((size_t)(2 * B_ + b)) * NP_ * INTER_;
    const unsigned short* A = theta + (size_t)w * 16 * INTER_;
    v8f acc[4] = {};
    for (int k = 0; k < INTER_; k += 32) {
        const v16bf a = load_a_bf16(A, INTER_, k);
#pragma unroll
        for (int s = 0; s < 4; ++s)
            acc[s] = wmma_bf16(a, load_b_nt_bf16(phi + (size_t)s * 16 * INTER_, INTER_, k), acc[s]);
    }
    const int lane = threadIdx.x & 31;
    const int col = lane & 15, hi = lane >> 4;
#pragma unroll
    for (int s = 0; s < 4; ++s)
#pragma unroll
        for (int v = 0; v < 8; ++v)
            sc[w * 16 + v + 8 * hi][s * 16 + col] = acc[s][v];
    __syncthreads();
    const int n = threadIdx.x;
    if (n < NP_) {
        unsigned short* arow = attn + ((size_t)b * NP_ + n) * NP_;
        if (n < N_) {
            float mx = -1e30f;
            for (int m = 0; m < N_; ++m) mx = fmaxf(mx, sc[n][m]);
            float sum = 0.f;
            for (int m = 0; m < N_; ++m) sum += __expf(sc[n][m] - mx);
            const float inv = 1.f / sum;
            for (int m = 0; m < N_; ++m) arow[m] = f32_to_bf16(__expf(sc[n][m] - mx) * inv);
            for (int m = N_; m < NP_; ++m) arow[m] = 0;
        } else {
            for (int m = 0; m < NP_; ++m) arow[m] = 0;
        }
    }
}

// mapT[b][n][o] = sum_m attn[n][m] * g_x[m][o]  (small K=64; strided B gather)
__global__ void __launch_bounds__(128) k_map(const unsigned short* __restrict__ attn,
                                             const unsigned short* __restrict__ proj,
                                             unsigned short* __restrict__ mapT) {
    __shared__ __align__(16) unsigned short stile[4][16][64];
    const int w  = threadIdx.x >> 5;
    const int wg = blockIdx.x * 4 + w;
    const int cg = wg & 15;
    const int rt = (wg >> 4) & 3;
    const int b  = wg >> 6;
    const unsigned short* A = attn + ((size_t)b * NP_ + rt * 16) * NP_;
    const unsigned short* G = proj + ((size_t)(0 * B_ + b)) * NP_ * INTER_ + cg * 64;  // g_x
    v8f acc[4] = {};
#pragma unroll
    for (int k = 0; k < NP_; k += 32) {
        const v16bf a = load_a_bf16(A, NP_, k);
#pragma unroll
        for (int s = 0; s < 4; ++s)
            acc[s] = wmma_bf16(a, load_b_strided_bf16(G + s * 16, INTER_, k), acc[s]);
    }
    unsigned short* dst = mapT + ((size_t)b * NP_ + rt * 16) * INTER_ + cg * 64;
    epilogue_store_bf16(stile[w], acc, nullptr, dst, INTER_);
}

// Spatial FC, split-K (deterministic): partial[kc][b][n] for kc in 0..48 (K chunk = 2048)
__global__ void __launch_bounds__(32) k_fc_partial(const float* __restrict__ x,
                                                   const float* __restrict__ fc_w,
                                                   float* __restrict__ partial) {
    const int wg = blockIdx.x;
    const int kc = wg % N_;
    const int rt = wg / N_;
    const int k0base = kc * C_;
    const float* A = x + (size_t)rt * 16 * KFLAT_;   // rows are batch indices; K-order = c*49+n
    v8f acc[4] = {};
    for (int k = 0; k < C_; k += 32) {
        const v16bf a = load_a_f32(A, KFLAT_, k0base + k);
#pragma unroll
        for (int s = 0; s < 4; ++s)
            acc[s] = wmma_bf16(a, load_b_nt_f32_guard(fc_w, (size_t)KFLAT_, k0base + k, s * 16, N_), acc[s]);
    }
    const int lane = threadIdx.x & 31;
    const int col = lane & 15, hi = lane >> 4;
    float* pout = partial + ((size_t)kc * B_ + rt * 16) * NP_;
#pragma unroll
    for (int s = 0; s < 4; ++s) {
        const int n = s * 16 + col;
#pragma unroll
        for (int v = 0; v < 8; ++v)
            pout[(size_t)(v + 8 * hi) * NP_ + n] = acc[s][v];
    }
}

__global__ void k_fc_reduce(const float* __restrict__ partial, const float* __restrict__ fc_b,
                            float* __restrict__ spatial) {
    const int i = blockIdx.x * blockDim.x + threadIdx.x;
    if (i >= B_ * N_) return;
    const int b = i / N_, n = i % N_;
    float s = fc_b[n];
    for (int kc = 0; kc < N_; ++kc) s += partial[((size_t)kc * B_ + b) * NP_ + n];
    spatial[i] = s;
}

// mask[c][n] = W_w[c][:] . mapT[n][:]  (NT), 64x64 per-wave block,
// fused epilogue: out = x + spatial[n]*mask
__global__ void __launch_bounds__(128) k_mask_out(const unsigned short* __restrict__ ww,
                                                  const unsigned short* __restrict__ mapT,
                                                  const float* __restrict__ spatial,
                                                  const float* __restrict__ x,
                                                  float* __restrict__ out) {
    __shared__ __align__(16) float ftile[4][64][65];   // ~66.6 KB
    const int w  = threadIdx.x >> 5;
    const int wg = blockIdx.x * 4 + w;
    const int ct = wg & 31;                 // 32 groups of 64 C-rows
    const int b  = wg >> 5;
    const unsigned short* A  = ww + (size_t)ct * 64 * INTER_;
    const unsigned short* Bm = mapT + (size_t)b * NP_ * INTER_;
    v8f acc[4][4] = {};
    for (int k = 0; k < INTER_; k += 32) {
        v16bf a[4];
#pragma unroll
        for (int r = 0; r < 4; ++r) a[r] = load_a_bf16(A + (size_t)r * 16 * INTER_, INTER_, k);
#pragma unroll
        for (int s = 0; s < 4; ++s) {
            const v16bf bf = load_b_nt_bf16(Bm + (size_t)s * 16 * INTER_, INTER_, k);
#pragma unroll
            for (int r = 0; r < 4; ++r)
                acc[r][s] = wmma_bf16(a[r], bf, acc[r][s]);
        }
    }
    const int lane = threadIdx.x & 31;
    const int col = lane & 15, hi = lane >> 4;
    float (*ft)[65] = ftile[w];
#pragma unroll
    for (int s = 0; s < 4; ++s)
#pragma unroll
        for (int r = 0; r < 4; ++r)
#pragma unroll
            for (int v = 0; v < 8; ++v)
                ft[r * 16 + v + 8 * hi][s * 16 + col] = acc[r][s][v];
    asm volatile("s_wait_dscnt 0" ::: "memory");
    const float* sp = spatial + (size_t)b * N_;
    for (int idx = lane; idx < 64 * N_; idx += 32) {
        const int r = idx / N_, n = idx % N_;
        const size_t o = ((size_t)b * C_ + ct * 64 + r) * N_ + n;
        out[o] = x[o] + sp[n] * ft[r][n];
    }
}

extern "C" void kernel_launch(void* const* d_in, const int* in_sizes, int n_in,
                              void* d_out, int out_size, void* d_ws, size_t ws_size,
                              hipStream_t stream) {
    const float* x    = (const float*)d_in[0];
    const float* g_w  = (const float*)d_in[1];
    const float* g_b  = (const float*)d_in[2];
    const float* th_w = (const float*)d_in[3];
    const float* th_b = (const float*)d_in[4];
    const float* ph_w = (const float*)d_in[5];
    const float* ph_b = (const float*)d_in[6];
    const float* W_w  = (const float*)d_in[7];
    const float* fc_w = (const float*)d_in[8];
    const float* fc_b = (const float*)d_in[9];
    float* out = (float*)d_out;
    (void)in_sizes; (void)n_in; (void)out_size; (void)ws_size;

    char* ws = (char*)d_ws;
    size_t off = 0;
    auto take = [&](size_t bytes) -> char* {
        char* p = ws + off;
        off += (bytes + 255) & ~(size_t)255;
        return p;
    };
    unsigned short* xT      = (unsigned short*)take((size_t)B_ * NP_ * C_ * 2);         // 64 MB
    unsigned short* w3      = (unsigned short*)take((size_t)3 * INTER_ * C_ * 2);       // 12.6 MB
    unsigned short* wwb     = (unsigned short*)take((size_t)C_ * INTER_ * 2);           // 4.2 MB
    unsigned short* proj    = (unsigned short*)take((size_t)3 * B_ * NP_ * INTER_ * 2); // 96 MB
    unsigned short* attn    = (unsigned short*)take((size_t)B_ * NP_ * NP_ * 2);        // 2 MB
    unsigned short* mapT    = (unsigned short*)take((size_t)B_ * NP_ * INTER_ * 2);     // 32 MB
    float*          partial = (float*)take((size_t)N_ * B_ * NP_ * 4);                  // 3.2 MB
    float*          spatial = (float*)take((size_t)B_ * N_ * 4);

    {
        const int n = INTER_ * C_;
        const int bl = 256, gr = (n + bl - 1) / bl;
        k_convert<<<gr, bl, 0, stream>>>(g_w,  w3, n);
        k_convert<<<gr, bl, 0, stream>>>(th_w, w3 + (size_t)INTER_ * C_, n);
        k_convert<<<gr, bl, 0, stream>>>(ph_w, w3 + (size_t)2 * INTER_ * C_, n);
        k_convert<<<gr, bl, 0, stream>>>(W_w,  wwb, n);   // C_*INTER_ == INTER_*C_
    }
    k_transpose_x<<<dim3(C_ / 64, B_), 256, 0, stream>>>(x, xT);
    k_proj<<<B_ * 3 * 16 / 4, 128, 0, stream>>>(xT, w3, g_b, th_b, ph_b, proj);
    k_scores_softmax<<<B_, 128, 0, stream>>>(proj, attn);
    k_map<<<B_ * 4 * 16 / 4, 128, 0, stream>>>(attn, proj, mapT);
    k_fc_partial<<<16 * N_, 32, 0, stream>>>(x, fc_w, partial);
    k_fc_reduce<<<(B_ * N_ + 255) / 256, 256, 0, stream>>>(partial, fc_b, spatial);
    k_mask_out<<<B_ * 32 / 4, 128, 0, stream>>>(wwb, mapT, spatial, x, out);
}
